// MultiHeadAttention_927712936184
// MI455X (gfx1250) — matrix operand order repaired and checked by reading, not compiled, not run on hardware
//
#include <hip/hip_runtime.h>

#define EMB   768
#define NTOK  2048
#define BATCH 4
#define HEADS 8
#define HDIM  96
#define SCALING 0.10206207261596575f  /* 96^-0.5 */

typedef __bf16        v16bf __attribute__((ext_vector_type(16)));
typedef __bf16        v8bf  __attribute__((ext_vector_type(8)));
typedef float         v8f   __attribute__((ext_vector_type(8)));
typedef unsigned int  v8u   __attribute__((ext_vector_type(8)));

__device__ __forceinline__ v16bf as_bf(v8u u) {
  return __builtin_bit_cast(v16bf, u);
}
__device__ __forceinline__ unsigned short bf1(float f) {
  return __builtin_bit_cast(unsigned short, (__bf16)f);
}
__device__ __forceinline__ v8f wmma_bf16(v16bf a, v16bf b, v8f c) {
  return __builtin_amdgcn_wmma_f32_16x16x32_bf16(false, a, false, b, (short)0, c,
                                                 false, false);
}

// ---- fragment loaders (bf16 sources only; all b128) -----------------------
// A-frag (16x32, 16-bit): lane holds row M=lane&15; element e -> k =
// e + (e>=8 ? 8 : 0) + half*8  => two contiguous 8-element runs.
// B-frag (32x16, 16-bit): lane holds col N=lane&15; element e -> k =
// e + (e>=8 ? 8 : 0) + half*8  => two contiguous 8-element runs.
__device__ __forceinline__ v16bf load_a_bf(const unsigned short* row, int k0, int half) {
  const unsigned short* p = row + k0 + half * 8;
  uint4 a = *(const uint4*)(p);
  uint4 b = *(const uint4*)(p + 16);
  v8u u = {a.x, a.y, a.z, a.w, b.x, b.y, b.z, b.w};
  return as_bf(u);
}
__device__ __forceinline__ v16bf load_b_bf(const unsigned short* row, int k0, int half) {
  const unsigned short* p = row + k0 + half * 8;
  uint4 a = *(const uint4*)(p);
  uint4 b = *(const uint4*)(p + 16);
  v8u u = {a.x, a.y, a.z, a.w, b.x, b.y, b.z, b.w};
  return as_bf(u);
}

// ---------------------------------------------------------------------------
// Kernel 0: one-shot f32 -> bf16 streaming convert (8 elems/lane, b256 in,
// b128 out, v_cvt_pk_bf16_f32 x4).  Removes all convert work + f32 operand
// traffic from the GEMM loops.
// ---------------------------------------------------------------------------
__global__ void __launch_bounds__(256) cvt_to_bf16(
    const float* __restrict__ s, unsigned short* __restrict__ d)
{
  const size_t i = ((size_t)blockIdx.x * 256 + threadIdx.x) * 8;
  v8f f = *(const v8f*)(s + i);
  v8bf b = __builtin_convertvector(f, v8bf);
  *(uint4*)(d + i) = __builtin_bit_cast(uint4, b);
}
__global__ void __launch_bounds__(256) cvt_w_bf16(
    const float* __restrict__ w0, const float* __restrict__ w1,
    const float* __restrict__ w2, const float* __restrict__ w3,
    unsigned short* __restrict__ d)
{
  const int which = blockIdx.y;
  const float* s = (which == 0) ? w0 : (which == 1) ? w1 : (which == 2) ? w2 : w3;
  const size_t i = ((size_t)blockIdx.x * 256 + threadIdx.x) * 8;
  v8f f = *(const v8f*)(s + i);
  v8bf b = __builtin_convertvector(f, v8bf);
  *(uint4*)(d + (size_t)which * EMB * EMB + i) = __builtin_bit_cast(uint4, b);
}

// ---------------------------------------------------------------------------
// Kernel 1: Q/K/V projection, pure bf16.  One wave per 16x64 output tile
// (A-frag reused across 4 B tiles): 10 b128 loads + 4 WMMAs per K-step.
// Q,K stored bf16 [B,H,N,96]; V stored TRANSPOSED [B,H,96,N] by swapping
// A/B roles (Vt = Wv * x^T) so GEMM loads and stores stay contiguous and
// attention reads V B-fragments as b128.
// ---------------------------------------------------------------------------
__global__ void __launch_bounds__(32) mha_qkv_proj(
    const unsigned short* __restrict__ xbf,
    const unsigned short* __restrict__ Wbf,   // [4][768][768], 0=q 1=k 2=v
    const float* __restrict__ bq, const float* __restrict__ bk,
    const float* __restrict__ bv,
    unsigned short* __restrict__ ws)
{
  const int lane = threadIdx.x;
  const int half = lane >> 4;
  const int nl   = lane & 15;
  const int which = blockIdx.z;

  const unsigned short* W = Wbf + (size_t)which * EMB * EMB;
  const float* bias = (which == 0) ? bq : (which == 1) ? bk : bv;
  unsigned short* out = ws + (size_t)which * ((size_t)BATCH * HEADS * NTOK * HDIM);

  int arow0, bcol0;
  const unsigned short *abase, *bbase;
  if (which < 2) {                       // rows = tokens, cols = out features
    arow0 = blockIdx.x * 16;
    bcol0 = blockIdx.y * 64;
    abase = xbf; bbase = W;
  } else {                               // V^T: rows = Wv rows, cols = tokens
    const int id = blockIdx.x * 12 + blockIdx.y;   // 6144 = 48 * 128
    arow0 = (id % 48) * 16;
    bcol0 = (id / 48) * 64;
    abase = W; bbase = xbf;
  }

  const unsigned short* arow = abase + (size_t)(arow0 + nl) * EMB;
  const unsigned short* brow = bbase + (size_t)(bcol0 + nl) * EMB;

  v8f acc[4];
#pragma unroll
  for (int g = 0; g < 4; ++g) acc[g] = (v8f){0.f,0.f,0.f,0.f,0.f,0.f,0.f,0.f};

  for (int kc = 0; kc < EMB / 32; ++kc) {
    const int k0 = kc * 32;
    const v16bf af = load_a_bf(arow, k0, half);
#pragma unroll
    for (int g = 0; g < 4; ++g) {
      const v16bf bf = load_b_bf(brow + (size_t)g * 16 * EMB, k0, half);
      acc[g] = wmma_bf16(af, bf, acc[g]);
    }
  }

  if (which < 2) {
    const int b  = arow0 / NTOK;         // tile never straddles a batch
    const int n0 = arow0 % NTOK + half * 8;
#pragma unroll
    for (int g = 0; g < 4; ++g) {
      const int colg = bcol0 + g * 16;   // 16-tile never straddles a head
      const int h = colg / HDIM;
      const int d = colg % HDIM + nl;
      const float bval = bias[colg + nl];
      unsigned short* orow =
          out + (((size_t)b * HEADS + h) * NTOK + n0) * HDIM + d;
#pragma unroll
      for (int r = 0; r < 8; ++r)
        orow[(size_t)r * HDIM] = bf1(acc[g][r] + bval);
    }
  } else {
    const int hh = arow0 / HDIM;         // 16 rows stay inside one head
    const int d0 = arow0 % HDIM + half * 8;
#pragma unroll
    for (int g = 0; g < 4; ++g) {
      const int tokg = bcol0 + g * 16;   // 16-tile never straddles a batch
      const int b = tokg / NTOK;
      const int n = tokg % NTOK + nl;
      unsigned short* orow =
          out + (((size_t)b * HEADS + hh) * HDIM + d0) * NTOK + n;
#pragma unroll
      for (int r = 0; r < 8; ++r)
        orow[(size_t)r * NTOK] = bf1(acc[g][r] + bias[arow0 + half * 8 + r]);
    }
  }
}

// ---------------------------------------------------------------------------
// Kernel 2: flash attention.  One wave per (b, h, 16-query tile); online
// softmax over 64 steps of 32 keys; 12 WMMAs per step.  V read from the
// transposed layout as contiguous b128 B-fragments; only the P relayout
// (C-layout -> A-layout) goes through 1KB of LDS.
// ---------------------------------------------------------------------------
__global__ void __launch_bounds__(32) mha_attention(
    const unsigned short* __restrict__ Qws,
    const unsigned short* __restrict__ Kws,
    const unsigned short* __restrict__ VTws,   // [B,H,96,N]
    unsigned short* __restrict__ Ows)
{
  __shared__ __align__(16) unsigned short Pt[16 * 32];

  const int lane = threadIdx.x;
  const int half = lane >> 4;
  const int nl   = lane & 15;
  const int q0   = blockIdx.x * 16;
  const int h    = blockIdx.y;
  const int b    = blockIdx.z;
  const size_t baseQK = ((size_t)b * HEADS + h) * NTOK * HDIM;
  const size_t baseVT = ((size_t)b * HEADS + h) * HDIM;

  v16bf qa[3];
  {
    const unsigned short* qrow = Qws + baseQK + (size_t)(q0 + nl) * HDIM;
#pragma unroll
    for (int c = 0; c < 3; ++c) qa[c] = load_a_bf(qrow, c * 32, half);
  }

  float mrow[8], lrow[8];
#pragma unroll
  for (int r = 0; r < 8; ++r) { mrow[r] = -3.0e38f; lrow[r] = 0.f; }
  v8f acc[6];
#pragma unroll
  for (int t = 0; t < 6; ++t) acc[t] = (v8f){0.f,0.f,0.f,0.f,0.f,0.f,0.f,0.f};

  for (int j = 0; j < NTOK / 32; ++j) {
    const int k0 = j * 32;
    const unsigned short* krow0 = Kws + baseQK + (size_t)(k0 + nl) * HDIM;
    const unsigned short* krow1 = krow0 + 16 * HDIM;

    // prefetch next step's K rows (global_prefetch_b8)
    if (j + 1 < NTOK / 32) {
      __builtin_prefetch(krow0 + 32 * HDIM, 0, 1);
      __builtin_prefetch(krow1 + 32 * HDIM, 0, 1);
    }

    // S = Q K^T : two key halves x three 32-wide d-chunks
    v8f s0 = (v8f){0.f,0.f,0.f,0.f,0.f,0.f,0.f,0.f};
    v8f s1 = (v8f){0.f,0.f,0.f,0.f,0.f,0.f,0.f,0.f};
#pragma unroll
    for (int c = 0; c < 3; ++c) {
      s0 = wmma_bf16(qa[c], load_b_bf(krow0, c * 32, half), s0);
      s1 = wmma_bf16(qa[c], load_b_bf(krow1, c * 32, half), s1);
    }

    // online softmax; C-layout rows live in VGPRs, cols in 16-lane groups
#pragma unroll
    for (int r = 0; r < 8; ++r) {
      float e0 = s0[r] * SCALING, e1 = s1[r] * SCALING;
      float mx = fmaxf(e0, e1);
      mx = fmaxf(mx, __shfl_xor(mx, 8, 16));
      mx = fmaxf(mx, __shfl_xor(mx, 4, 16));
      mx = fmaxf(mx, __shfl_xor(mx, 2, 16));
      mx = fmaxf(mx, __shfl_xor(mx, 1, 16));
      const float mnew = fmaxf(mrow[r], mx);
      const float corr = __expf(mrow[r] - mnew);
      const float p0 = __expf(e0 - mnew);
      const float p1 = __expf(e1 - mnew);
      float rs = p0 + p1;
      rs += __shfl_xor(rs, 8, 16);
      rs += __shfl_xor(rs, 4, 16);
      rs += __shfl_xor(rs, 2, 16);
      rs += __shfl_xor(rs, 1, 16);
      lrow[r] = lrow[r] * corr + rs;
      mrow[r] = mnew;
#pragma unroll
      for (int t = 0; t < 6; ++t) acc[t][r] *= corr;
      const int m = r + half * 8;
      Pt[m * 32 + nl]      = bf1(p0);
      Pt[m * 32 + 16 + nl] = bf1(p1);
    }

    // P (16x32) back out of LDS as an A-fragment
    const v16bf pa = load_a_bf(Pt + nl * 32, 0, half);

    // acc += P @ V : V^T rows are d-columns -> contiguous b128 B-fragments
#pragma unroll
    for (int t = 0; t < 6; ++t) {
      const unsigned short* vrow =
          VTws + (baseVT + (size_t)(t * 16 + nl)) * NTOK + k0;
      if (j + 1 < NTOK / 32) __builtin_prefetch(vrow + 32, 0, 1);
      acc[t] = wmma_bf16(pa, load_b_bf(vrow, 0, half), acc[t]);
    }
  }

  // normalize and store bf16 O[B, N, EMB]
#pragma unroll
  for (int t = 0; t < 6; ++t) {
#pragma unroll
    for (int r = 0; r < 8; ++r) {
      const int m = r + half * 8;
      const float o = acc[t][r] / lrow[r];
      Ows[((size_t)b * NTOK + q0 + m) * EMB + h * HDIM + t * 16 + nl] = bf1(o);
    }
  }
}

// ---------------------------------------------------------------------------
// Kernel 3: output projection.  out = O @ Wo^T + bo (f32).  16x64 per wave,
// pure bf16 operands.
// ---------------------------------------------------------------------------
__global__ void __launch_bounds__(32) mha_out_proj(
    const unsigned short* __restrict__ Ows,
    const unsigned short* __restrict__ Wobf,
    const float* __restrict__ bo,
    float* __restrict__ out)
{
  const int lane = threadIdx.x;
  const int half = lane >> 4;
  const int nl   = lane & 15;
  const int tok0 = blockIdx.x * 16;
  const int col0 = blockIdx.y * 64;

  const unsigned short* orow = Ows + (size_t)(tok0 + nl) * EMB;
  const unsigned short* wrow = Wobf + (size_t)(col0 + nl) * EMB;

  v8f acc[4];
#pragma unroll
  for (int g = 0; g < 4; ++g) acc[g] = (v8f){0.f,0.f,0.f,0.f,0.f,0.f,0.f,0.f};

  for (int kc = 0; kc < EMB / 32; ++kc) {
    const int k0 = kc * 32;
    const v16bf af = load_a_bf(orow, k0, half);
#pragma unroll
    for (int g = 0; g < 4; ++g) {
      const v16bf bf = load_b_bf(wrow + (size_t)g * 16 * EMB, k0, half);
      acc[g] = wmma_bf16(af, bf, acc[g]);
    }
  }

#pragma unroll
  for (int g = 0; g < 4; ++g) {
    const float bval = bo[col0 + g * 16 + nl];
    float* optr = out + (size_t)(tok0 + half * 8) * EMB + col0 + g * 16 + nl;
#pragma unroll
    for (int r = 0; r < 8; ++r)
      optr[(size_t)r * EMB] = acc[g][r] + bval;
  }
}

// ---------------------------------------------------------------------------
extern "C" void kernel_launch(void* const* d_in, const int* in_sizes, int n_in,
                              void* d_out, int out_size, void* d_ws, size_t ws_size,
                              hipStream_t stream) {
  const float* x  = (const float*)d_in[0];
  const float* Wq = (const float*)d_in[1];
  const float* bq = (const float*)d_in[2];
  const float* Wk = (const float*)d_in[3];
  const float* bk = (const float*)d_in[4];
  const float* Wv = (const float*)d_in[5];
  const float* bv = (const float*)d_in[6];
  const float* Wo = (const float*)d_in[7];
  const float* bo = (const float*)d_in[8];
  float* out = (float*)d_out;

  unsigned short* ws = (unsigned short*)d_ws;
  const size_t seg = (size_t)BATCH * HEADS * NTOK * HDIM;  // 6,291,456 elems
  unsigned short* Qws  = ws;
  unsigned short* Kws  = ws + seg;
  unsigned short* VTws = ws + 2 * seg;        // V transposed [B,H,96,N]
  unsigned short* Ows  = ws + 3 * seg;
  unsigned short* xbf  = ws + 4 * seg;        // x in bf16 [8192, 768]
  unsigned short* Wbf  = ws + 5 * seg;        // [4][768][768] bf16 (q,k,v,o)
  // total ws use: (5*seg + 4*EMB*EMB)*2 B ~= 68 MB

  // Stage 0: one-shot f32 -> bf16 converts
  cvt_to_bf16<<<dim3((unsigned)(seg / 8 / 256)), 256, 0, stream>>>(x, xbf);
  cvt_w_bf16<<<dim3(EMB * EMB / 8 / 256, 4), 256, 0, stream>>>(Wq, Wk, Wv, Wo, Wbf);

  // Stage 1: QKV projection (pure bf16 GEMMs)
  dim3 gA((BATCH * NTOK) / 16, EMB / 64, 3);   // (512, 12, 3)
  mha_qkv_proj<<<gA, 32, 0, stream>>>(xbf, Wbf, bq, bk, bv, ws);

  // Stage 2: attention
  dim3 gB(NTOK / 16, HEADS, BATCH);            // (128, 8, 4)
  mha_attention<<<gB, 32, 0, stream>>>(Qws, Kws, VTws, Ows);

  // Stage 3: output projection
  dim3 gC((BATCH * NTOK) / 16, EMB / 64);      // (512, 12)
  mha_out_proj<<<gC, 32, 0, stream>>>(Ows, Wbf + (size_t)3 * EMB * EMB, bo, out);
}
